// SimpleMetaNet_19241453486389
// MI455X (gfx1250) — matrix operand order repaired and checked by reading, not compiled, not run on hardware
//
#include <hip/hip_runtime.h>
#include <hip/hip_bf16.h>

// ---------------- CDNA5 WMMA plumbing (gfx1250, wave32) ----------------
typedef _Float16 f16;
typedef __attribute__((ext_vector_type(16))) _Float16 v16h;
typedef __attribute__((ext_vector_type(8)))  _Float16 v8h;
typedef __attribute__((ext_vector_type(8)))  float    v8f;

// A fragment (16x32 f16). Row-major source, stride in f16 elements.
// Lane l: row = l&15, khalf = l>>4. Elems 0..15 -> K = khalf*16 + i
// (contiguous, matching load_b_frag).
__device__ inline v16h load_a_frag(const f16* base, int stride) {
    int lane = threadIdx.x & 31;
    int r = lane & 15, kh = lane >> 4;
    const f16* p = base + r * stride + kh * 16;
    v8h lo = *(const v8h*)p;
    v8h hi = *(const v8h*)(p + 8);
    v16h a;
    #pragma unroll
    for (int i = 0; i < 8; ++i) { a[i] = lo[i]; a[8 + i] = hi[i]; }
    return a;
}

// B fragment (32x16 f16) from row-major W[n][k] (global, f16).
// Lane l: col n = l&15, khalf = l>>4; 16 contiguous K values per lane.
__device__ inline v16h load_b_frag(const f16* W, int n0, int Kd, int k0) {
    int lane = threadIdx.x & 31;
    int n = lane & 15, kh = lane >> 4;
    return *(const v16h*)(W + (size_t)(n0 + n) * Kd + k0 + kh * 16);
}

__device__ inline v8f wmma16(v16h a, v16h b, v8f c) {
    return __builtin_amdgcn_wmma_f32_16x16x32_f16(
        /*neg_a=*/false, a, /*neg_b=*/false, b,
        /*c_mod=*/(short)0, c, /*reuse_a=*/false, /*reuse_b=*/false);
}

// Async global->LDS copy, 4 bytes per lane (ASYNCcnt-tracked, no VGPR return).
// x rows are 129 floats (516B stride) so only B32 is alignment-safe.
__device__ inline void async_copy_b32(unsigned int lds_off, const void* gaddr) {
    asm volatile("global_load_async_to_lds_b32 %0, %1, off"
                 :: "v"(lds_off), "v"((unsigned long long)(uintptr_t)gaddr)
                 : "memory");
}
__device__ inline void wait_async_all() {
    asm volatile("s_wait_asynccnt 0x0" ::: "memory");
}
// Generic LDS pointer -> LDS byte offset (flat LDS aperture: addr[31:0] = offset).
__device__ inline unsigned int lds_offset(const void* p) {
    return (unsigned int)(uintptr_t)p;
}

#define B_   256
#define M_   2048
#define D_   128
#define H_   256
#define MSG_ 256
#define K_   64
#define OUT_ 64

// ---------------- small utility kernels ----------------
__global__ void convert_kernel(const float* __restrict__ s, f16* __restrict__ d, int n) {
    int i = blockIdx.x * 256 + threadIdx.x;
    if (i < n) d[i] = (f16)s[i];
}
__global__ void zero_kernel(float* p, int n) {
    int i = blockIdx.x * 256 + threadIdx.x;
    if (i < n) p[i] = 0.f;
}
__global__ void add_kernel(const float* a, const float* b, float* c, int n) {
    int i = blockIdx.x * 256 + threadIdx.x;
    if (i < n) c[i] = a[i] + b[i];
}
// cat = [xt2 (256) | msg = xt_post[:,2048:2304]]
__global__ void cat_kernel(const float* __restrict__ xt2, const float* __restrict__ xt_post,
                           float* __restrict__ cat) {
    int i = blockIdx.x * 256 + threadIdx.x;
    if (i < B_ * (H_ + MSG_)) {
        int b = i >> 9, j = i & 511;
        cat[i] = (j < H_) ? xt2[b * H_ + j]
                          : xt_post[(size_t)b * (M_ + MSG_) + M_ + (j - H_)];
    }
}
// per-batch sum of labels (x[...,128])
__global__ void labelsum_kernel(const float* __restrict__ x, float* __restrict__ lsum) {
    __shared__ float s[256];
    int b = blockIdx.x, tid = threadIdx.x;
    float acc = 0.f;
    for (int m = tid; m < M_; m += 256)
        acc += x[((size_t)b * M_ + m) * (D_ + 1) + D_];
    s[tid] = acc; __syncthreads();
    for (int st = 128; st > 0; st >>= 1) { if (tid < st) s[tid] += s[tid + st]; __syncthreads(); }
    if (tid == 0) lsum[b] = s[0];
}
// c0 = relu(k2_b0) @ k2_w1^T + k2_b1  (response of an all-zero row in pass 2)
__global__ void c0_kernel(const float* __restrict__ b0, const f16* __restrict__ W1,
                          const float* __restrict__ b1, float* __restrict__ c0) {
    int h = threadIdx.x;  // 256 threads
    float s = 0.f;
    for (int j = 0; j < H_; ++j) {
        float r = b0[j]; r = r > 0.f ? r : 0.f;
        s += r * (float)W1[(size_t)h * H_ + j];
    }
    c0[h] = s + b1[h];
}

// ---------------- dominant fused pass-1 kernel ----------------
// One block = one (batch, 128-row m-tile). 8 waves, each owns a 16-row stripe.
// feats f32 async-staged into LDS (overlaid on the H region), converted to f16,
// then GEMM1(K=128)+relu -> LDS H; GEMM2(K=256); label-weighted row reduction
// -> ds_add_f32 -> global_atomic_add_f32 into xt[b][h].
// A-fragments are hoisted into registers across the n-tile loop (16x fewer LDS reads).
__global__ __launch_bounds__(256) void pass1_kernel(
    const float* __restrict__ x, const f16* __restrict__ W0, const float* __restrict__ b0,
    const f16* __restrict__ W1, const float* __restrict__ b1, float* __restrict__ xt)
{
    extern __shared__ char smem1[];
    f16*   As  = (f16*)smem1;                 // 128 x 136 f16
    f16*   Hs  = As + 128 * 136;              // 128 x 264 f16
    float* raw = (float*)Hs;                  // 128 x 128 f32 staging, overlaid on Hs
    float* lab = (float*)(Hs + 128 * 264);    // 128
    float* xts = lab + 128;                   // 256

    int blk = blockIdx.x;
    int b = blk >> 4;
    int m0 = (blk & 15) * 128;
    int tid = threadIdx.x, lane = tid & 31, wave = tid >> 5;

    const float* xb = x + ((size_t)b * M_ + m0) * (D_ + 1);

    // async-stage the 64KB f32 feats tile straight into LDS
    unsigned int raw_base = lds_offset(raw);
    for (int e = tid; e < 128 * 128; e += 256) {
        int m = e >> 7, d = e & 127;   // consecutive lanes -> consecutive d (coalesced)
        async_copy_b32(raw_base + (unsigned int)e * 4u, xb + (size_t)m * (D_ + 1) + d);
    }
    // overlap label loads / accumulator init with the in-flight async copies
    if (tid < 128) lab[tid] = xb[(size_t)tid * (D_ + 1) + D_];
    xts[tid] = 0.f;
    wait_async_all();
    __syncthreads();

    // LDS f32 -> LDS f16 (padded stride 136 for conflict-free fragment reads)
    for (int e = tid; e < 128 * 128; e += 256) {
        int m = e >> 7, d = e & 127;
        As[m * 136 + d] = (f16)raw[e];
    }
    __syncthreads();

    int r0 = wave * 16, hi = lane >> 4, col = lane & 15;

    // GEMM1: H = relu(feats @ W0^T + b0), 128x256, K=128
    v16h a1[4];
    #pragma unroll
    for (int kt = 0; kt < 4; ++kt)
        a1[kt] = load_a_frag(As + r0 * 136 + kt * 32, 136);
    for (int nt = 0; nt < 16; ++nt) {
        v8f acc = {};
        #pragma unroll
        for (int kt = 0; kt < 4; ++kt)
            acc = wmma16(a1[kt], load_b_frag(W0, nt * 16, D_, kt * 32), acc);
        float bv = b0[nt * 16 + col];
        #pragma unroll
        for (int v = 0; v < 8; ++v) {
            float val = acc[v] + bv;
            val = val > 0.f ? val : 0.f;
            Hs[(r0 + v + 8 * hi) * 264 + nt * 16 + col] = (f16)val;
        }
    }
    __syncthreads();

    // GEMM2: (H @ W1^T + b1) * label, reduced over rows
    v16h a2[8];
    #pragma unroll
    for (int kt = 0; kt < 8; ++kt)
        a2[kt] = load_a_frag(Hs + r0 * 264 + kt * 32, 264);
    for (int nt = 0; nt < 16; ++nt) {
        v8f acc = {};
        #pragma unroll
        for (int kt = 0; kt < 8; ++kt)
            acc = wmma16(a2[kt], load_b_frag(W1, nt * 16, H_, kt * 32), acc);
        float bv = b1[nt * 16 + col];
        float part = 0.f;
        #pragma unroll
        for (int v = 0; v < 8; ++v)
            part += (acc[v] + bv) * lab[r0 + v + 8 * hi];
        atomicAdd(&xts[nt * 16 + col], part);        // ds_add_f32
    }
    __syncthreads();
    unsafeAtomicAdd(&xt[(size_t)b * H_ + tid], xts[tid] * (1.0f / (float)M_));
}

// ---------------- pass-2 on the 64 selected rows ----------------
__global__ __launch_bounds__(256) void pass2_kernel(
    const float* __restrict__ x, const int* __restrict__ idx,
    const f16* __restrict__ W0, const float* __restrict__ b0,
    const f16* __restrict__ W1, const float* __restrict__ b1,
    const float* __restrict__ lsum, const float* __restrict__ c0,
    float* __restrict__ xt2)
{
    __shared__ f16   As[64 * 136];
    __shared__ f16   Hs[64 * 264];
    __shared__ float lab[64];
    __shared__ float xts[256];
    __shared__ float sls;
    float* raw = (float*)Hs;   // 64 x 128 f32 staging, overlaid on Hs

    int b = blockIdx.x, tid = threadIdx.x, lane = tid & 31, wave = tid >> 5;
    if (tid == 0) sls = 0.f;
    xts[tid] = 0.f;
    __syncthreads();

    // async-gather the 64 selected rows into LDS
    unsigned int raw_base = lds_offset(raw);
    for (int e = tid; e < 64 * 128; e += 256) {
        int r = e >> 7, d = e & 127;
        int m = idx[b * K_ + r];
        async_copy_b32(raw_base + (unsigned int)e * 4u,
                       x + ((size_t)b * M_ + m) * (D_ + 1) + d);
    }
    if (tid < 64) {
        int m = idx[b * K_ + tid];
        float L = x[((size_t)b * M_ + m) * (D_ + 1) + D_];
        lab[tid] = L;
        atomicAdd(&sls, L);
    }
    wait_async_all();
    __syncthreads();
    for (int e = tid; e < 64 * 128; e += 256) {
        int r = e >> 7, d = e & 127;
        As[r * 136 + d] = (f16)raw[e];
    }
    __syncthreads();

    int rt = wave & 3, r0 = rt * 16, hi = lane >> 4, col = lane & 15;
    int ntb = (wave >> 2) * 8;

    v16h a1[4];
    #pragma unroll
    for (int kt = 0; kt < 4; ++kt)
        a1[kt] = load_a_frag(As + r0 * 136 + kt * 32, 136);
    for (int j = 0; j < 8; ++j) {
        int nt = ntb + j;
        v8f acc = {};
        #pragma unroll
        for (int kt = 0; kt < 4; ++kt)
            acc = wmma16(a1[kt], load_b_frag(W0, nt * 16, D_, kt * 32), acc);
        float bv = b0[nt * 16 + col];
        #pragma unroll
        for (int v = 0; v < 8; ++v) {
            float val = acc[v] + bv;
            val = val > 0.f ? val : 0.f;
            Hs[(r0 + v + 8 * hi) * 264 + nt * 16 + col] = (f16)val;
        }
    }
    __syncthreads();

    v16h a2[8];
    #pragma unroll
    for (int kt = 0; kt < 8; ++kt)
        a2[kt] = load_a_frag(Hs + r0 * 264 + kt * 32, 264);
    for (int j = 0; j < 8; ++j) {
        int nt = ntb + j;
        v8f acc = {};
        #pragma unroll
        for (int kt = 0; kt < 8; ++kt)
            acc = wmma16(a2[kt], load_b_frag(W1, nt * 16, H_, kt * 32), acc);
        float bv = b1[nt * 16 + col];
        float part = 0.f;
        #pragma unroll
        for (int v = 0; v < 8; ++v)
            part += (acc[v] + bv) * lab[r0 + v + 8 * hi];
        atomicAdd(&xts[nt * 16 + col], part);
    }
    __syncthreads();
    // masked-out rows contribute c0 * (sum_all_labels - sum_selected_labels)
    xt2[(size_t)b * H_ + tid] = (xts[tid] + (lsum[b] - sls) * c0[tid]) * (1.0f / (float)M_);
}

// ---------------- generic WMMA GEMM: C = act(A@W^T + bias), A f32, W f16 ----------------
__global__ __launch_bounds__(256) void gemm_kernel(
    const float* __restrict__ A, const f16* __restrict__ W,
    const float* __restrict__ bias, float* __restrict__ C,
    int N, int Kd, int relu)
{
    __shared__ f16 As[128 * 40];
    int tid = threadIdx.x, lane = tid & 31, wave = tid >> 5;
    int row0 = blockIdx.x * 128;
    int col0 = blockIdx.y * 128;
    int ntiles = (N - col0) < 128 ? (N - col0) / 16 : 8;
    v8f acc[8] = {};
    int r0 = wave * 16, hi = lane >> 4, col = lane & 15;

    for (int k0 = 0; k0 < Kd; k0 += 32) {
        for (int e = tid; e < 128 * 32; e += 256) {
            int m = e >> 5, k = e & 31;
            As[m * 40 + k] = (f16)A[(size_t)(row0 + m) * Kd + k0 + k];
        }
        __syncthreads();
        v16h a = load_a_frag(As + r0 * 40, 40);
        for (int nt = 0; nt < ntiles; ++nt) {
            v16h bf = load_b_frag(W, col0 + nt * 16, Kd, k0);
            acc[nt] = wmma16(a, bf, acc[nt]);
        }
        __syncthreads();
    }
    for (int nt = 0; nt < ntiles; ++nt) {
        int n = col0 + nt * 16 + col;
        float bv = bias ? bias[n] : 0.f;
        #pragma unroll
        for (int v = 0; v < 8; ++v) {
            float val = acc[nt][v] + bv;
            if (relu) val = val > 0.f ? val : 0.f;
            C[(size_t)(row0 + r0 + v + 8 * hi) * N + n] = val;
        }
    }
}

// ---------------- top-64 selection (softmax monotone: select on logits+gumbel) ----------------
__global__ __launch_bounds__(256) void topk_kernel(
    const float* __restrict__ xt_post, const float* __restrict__ gn, int* __restrict__ idx)
{
    __shared__ float zs[M_];
    __shared__ float smax[256];
    __shared__ int   sidx[256];
    int b = blockIdx.x, tid = threadIdx.x;
    for (int m = tid; m < M_; m += 256)
        zs[m] = xt_post[(size_t)b * (M_ + MSG_) + m] + gn[(size_t)b * M_ + m];
    __syncthreads();
    for (int it = 0; it < K_; ++it) {
        float best = -1e30f; int bi = 0;
        for (int m = tid; m < M_; m += 256)
            if (zs[m] > best) { best = zs[m]; bi = m; }
        smax[tid] = best; sidx[tid] = bi;
        __syncthreads();
        for (int s = 128; s > 0; s >>= 1) {
            if (tid < s && smax[tid + s] > smax[tid]) { smax[tid] = smax[tid + s]; sidx[tid] = sidx[tid + s]; }
            __syncthreads();
        }
        if (tid == 0) { idx[b * K_ + it] = sidx[0]; zs[sidx[0]] = -1e30f; }
        __syncthreads();
    }
}

// ---------------- host ----------------
extern "C" void kernel_launch(void* const* d_in, const int* in_sizes, int n_in,
                              void* d_out, int out_size, void* d_ws, size_t ws_size,
                              hipStream_t stream)
{
    (void)in_sizes; (void)n_in; (void)out_size; (void)ws_size; // needs ~8 MB ws
    const float* x    = (const float*)d_in[0];
    const float* gn   = (const float*)d_in[1];
    const float* k1w0 = (const float*)d_in[2];
    const float* k1b0 = (const float*)d_in[3];
    const float* k1w1 = (const float*)d_in[4];
    const float* k1b1 = (const float*)d_in[5];
    const float* k2w0 = (const float*)d_in[6];
    const float* k2b0 = (const float*)d_in[7];
    const float* k2w1 = (const float*)d_in[8];
    const float* k2b1 = (const float*)d_in[9];
    const float* m1w0 = (const float*)d_in[10];
    const float* m1b0 = (const float*)d_in[11];
    const float* m1w1 = (const float*)d_in[12];
    const float* m1b1 = (const float*)d_in[13];
    const float* m2w0 = (const float*)d_in[14];
    const float* m2b0 = (const float*)d_in[15];
    const float* m2w1 = (const float*)d_in[16];
    const float* m2b1 = (const float*)d_in[17];
    const float* llw  = (const float*)d_in[18];
    const float* llb  = (const float*)d_in[19];
    float* out = (float*)d_out;

    char* ws = (char*)d_ws;
    auto alloc = [&](size_t bytes) -> void* {
        void* p = (void*)ws; ws += (bytes + 255) & ~(size_t)255; return p;
    };
    f16* h_k1w0 = (f16*)alloc((size_t)H_ * D_ * 2);
    f16* h_k1w1 = (f16*)alloc((size_t)H_ * H_ * 2);
    f16* h_k2w0 = (f16*)alloc((size_t)H_ * D_ * 2);
    f16* h_k2w1 = (f16*)alloc((size_t)H_ * H_ * 2);
    f16* h_m1w0 = (f16*)alloc((size_t)H_ * H_ * 2);
    f16* h_m1w1 = (f16*)alloc((size_t)(MSG_ + M_) * H_ * 2);
    f16* h_m2w0 = (f16*)alloc((size_t)H_ * (H_ + MSG_) * 2);
    f16* h_m2w1 = (f16*)alloc((size_t)H_ * H_ * 2);
    f16* h_llw  = (f16*)alloc((size_t)OUT_ * H_ * 2);
    float* xt      = (float*)alloc((size_t)B_ * H_ * 4);
    float* xt_mid  = (float*)alloc((size_t)B_ * H_ * 4);
    float* xt_post = (float*)alloc((size_t)B_ * (M_ + MSG_) * 4);
    float* xt2     = (float*)alloc((size_t)B_ * H_ * 4);
    float* lsum    = (float*)alloc((size_t)B_ * 4);
    float* c0buf   = (float*)alloc((size_t)H_ * 4);
    float* catb    = (float*)alloc((size_t)B_ * (H_ + MSG_) * 4);
    float* y1      = (float*)alloc((size_t)B_ * H_ * 4);
    float* y2      = (float*)alloc((size_t)B_ * H_ * 4);
    float* y3      = (float*)alloc((size_t)B_ * H_ * 4);
    int*   idxb    = (int*)alloc((size_t)B_ * K_ * 4);

    auto conv = [&](const float* s, f16* d, int n) {
        convert_kernel<<<(n + 255) / 256, 256, 0, stream>>>(s, d, n);
    };
    conv(k1w0, h_k1w0, H_ * D_);
    conv(k1w1, h_k1w1, H_ * H_);
    conv(k2w0, h_k2w0, H_ * D_);
    conv(k2w1, h_k2w1, H_ * H_);
    conv(m1w0, h_m1w0, H_ * H_);
    conv(m1w1, h_m1w1, (MSG_ + M_) * H_);
    conv(m2w0, h_m2w0, H_ * (H_ + MSG_));
    conv(m2w1, h_m2w1, H_ * H_);
    conv(llw,  h_llw,  OUT_ * H_);

    zero_kernel<<<(B_ * H_ + 255) / 256, 256, 0, stream>>>(xt, B_ * H_);
    c0_kernel<<<1, 256, 0, stream>>>(k2b0, h_k2w1, k2b1, c0buf);
    labelsum_kernel<<<B_, 256, 0, stream>>>(x, lsum);

    // pass 1: dominant WMMA work (103 GFLOP)
    size_t p1_lds = (size_t)(128 * 136 + 128 * 264) * 2 + (size_t)(128 + 256) * 4;
    pass1_kernel<<<B_ * (M_ / 128), 256, p1_lds, stream>>>(x, h_k1w0, k1b0, h_k1w1, k1b1, xt);

    // m1 MLP: xt_mid = relu(xt@m1w0^T+b); xt_post = xt_mid@m1w1^T+b
    gemm_kernel<<<dim3(2, 2),  256, 0, stream>>>(xt,     h_m1w0, m1b0, xt_mid,  H_,        H_, 1);
    gemm_kernel<<<dim3(2, 18), 256, 0, stream>>>(xt_mid, h_m1w1, m1b1, xt_post, M_ + MSG_, H_, 0);

    topk_kernel<<<B_, 256, 0, stream>>>(xt_post, gn, idxb);

    // pass 2: only the 64 selected rows per batch + c0 correction
    pass2_kernel<<<B_, 256, 0, stream>>>(x, idxb, h_k2w0, k2b0, h_k2w1, k2b1, lsum, c0buf, xt2);

    cat_kernel<<<(B_ * (H_ + MSG_) + 255) / 256, 256, 0, stream>>>(xt2, xt_post, catb);
    gemm_kernel<<<dim3(2, 2), 256, 0, stream>>>(catb, h_m2w0, m2b0, y1, H_, H_ + MSG_, 1);
    gemm_kernel<<<dim3(2, 2), 256, 0, stream>>>(y1,   h_m2w1, m2b1, y2, H_, H_,        1);
    add_kernel<<<(B_ * H_ + 255) / 256, 256, 0, stream>>>(y2, xt2, y3, B_ * H_);
    gemm_kernel<<<dim3(2, 1), 256, 0, stream>>>(y3, h_llw, llb, out, OUT_, H_, 0);
}
